// PoseRefineHead_25881472926454
// MI455X (gfx1250) — matrix operand order repaired and checked by reading, not compiled, not run on hardware
//
#include <hip/hip_runtime.h>

// ---------------------------------------------------------------------------
// MI455X / gfx1250 PoseRefineHead.
// Big GEMMs: v_wmma_f32_16x16x32_f16, f16 operands end-to-end.
// Tiles arrive via gfx1250 async global->LDS DMA (ASYNCcnt) into
// double-buffered LDS; weights are pre-swizzled N-major K-pair-packed so
// BOTH A and B fragments are two ds_load_b128 straight into WMMA operand
// registers (no register shuffling). One barrier per K-step.
// ---------------------------------------------------------------------------

typedef __attribute__((ext_vector_type(16))) _Float16   v16h;
typedef __attribute__((ext_vector_type(4)))  _Float16   v4h;
typedef __attribute__((ext_vector_type(8)))  float      v8f;
typedef __attribute__((ext_vector_type(4)))  unsigned   v4u;

union FragH {            // 16 f16 = 8 VGPRs = one WMMA A/B operand
  v16h h;
  v4u  q[2];
  unsigned u[8];
};

#define NPTS   8192
#define FDIM   512
#define KNN_K  32
#define BN_EPS 1e-5f

// ---- gfx1250 async global->LDS copy (16B per lane), tracked by ASYNCcnt ----
__device__ __forceinline__ void async_copy_b128(unsigned lds_addr, const void* gaddr) {
  asm volatile("global_load_async_to_lds_b128 %0, %1, off"
               :: "v"(lds_addr), "v"(gaddr) : "memory");
}
__device__ __forceinline__ void wait_async0() {
  asm volatile("s_wait_asynccnt 0x0" ::: "memory");
}
__device__ __forceinline__ unsigned lds_off(const void* p) {
  return (unsigned)(size_t)p;   // low 32 bits of generic LDS pointer = LDS byte offset
}

// ---------------------------------------------------------------------------
// Weight prep: f32 [K,N] -> f16, N-major K-pair-packed:
//   dst[(n*(K/2) + kp)*2 + {0,1}] = src[2kp*N + n], src[(2kp+1)*N + n]
// A B-tile is then a contiguous 2-D copy, and a B fragment is 32 contiguous
// bytes per lane (two ds_load_b128), mirroring the A fragment path.
// ---------------------------------------------------------------------------
__global__ __launch_bounds__(256) void cvt_pack_kernel(const float* __restrict__ s,
                                                       _Float16* __restrict__ d,
                                                       int K, int N) {
  int i = blockIdx.x * 256 + threadIdx.x;
  int K2 = K >> 1;
  int tot = K2 * N;
  if (i >= tot) return;
  int n = i / K2, kp = i - n * K2;
  d[2 * (size_t)i]     = (_Float16)s[(size_t)(2 * kp)     * N + n];
  d[2 * (size_t)i + 1] = (_Float16)s[(size_t)(2 * kp + 1) * N + n];
}

// ---------------------------------------------------------------------------
// Encoder layer 0: [N,3] @ [3,64] + bias, BN, ReLU -> f16 (K=3: plain VALU)
// ---------------------------------------------------------------------------
__global__ __launch_bounds__(256) void enc0_kernel(const float* __restrict__ pts,
                                                   const float* __restrict__ w,
                                                   const float* __restrict__ b,
                                                   const float* __restrict__ g,
                                                   const float* __restrict__ be,
                                                   _Float16* __restrict__ out) {
  int i = blockIdx.x * 4 + (threadIdx.x >> 6);
  int o = threadIdx.x & 63;
  float x = pts[i * 3 + 0], y = pts[i * 3 + 1], z = pts[i * 3 + 2];
  float v = x * w[o] + y * w[64 + o] + z * w[128 + o] + b[o];
  v = v * (g[o] * rsqrtf(1.f + BN_EPS)) + be[o];
  out[(size_t)i * 64 + o] = (_Float16)fmaxf(v, 0.f);
}

// ---------------------------------------------------------------------------
// WMMA GEMM: C[M,N] = A16[M,K] @ W + bias (+BN+ReLU) (+f16 out), templated
// on MODE (bit0: BN+ReLU fuse, bit1: f16 output) so the epilogue is
// branch-free. Workgroup 256 thr = 8 waves, tile 128x128; wave = 32x64 =
// 2x4 WMMA accumulators. Per K-step per wave: 4x async b128 issue,
// 12x ds_load_b128, 8x v_wmma, s_wait_asynccnt, barrier.
// ---------------------------------------------------------------------------
#define GBM 128
#define GBN 128
#define GBK 32

template <int MODE>
__global__ __launch_bounds__(256) void gemm_wmma_kernel(
    const _Float16* __restrict__ A, int lda,
    const _Float16* __restrict__ Wp,        // N-major pair-packed: [n][K/2] u32 cells
    const float* __restrict__ bias,
    const float* __restrict__ gamma,
    const float* __restrict__ beta,
    void* __restrict__ C, int ldc, int col_off,
    int N, int K) {
  __shared__ __align__(16) _Float16 As[2][GBM][40];   // 80B rows: 16B aligned, bank-spread
  __shared__ __align__(16) unsigned Bs[2][GBN][20];   // [n][kp] u32, 80B rows

  const int tid  = threadIdx.x;
  const int lane = tid & 31;
  const int wid  = tid >> 5;
  const int wm   = wid & 3;          // 4 M-quarters of 32 rows
  const int wn   = wid >> 2;         // 2 N-halves of 64 cols
  const int m0   = blockIdx.x * GBM;
  const int n0   = blockIdx.y * GBN;
  const int K2   = K >> 1;

  const int mrow  = lane & 15;
  const int khalf = lane >> 4;

  v8f acc[2][4] = {};

  auto stage = [&](int k0s, int nb) {
    #pragma unroll
    for (int i = 0; i < 2; ++i) {              // A: 128 rows x 4 chunks of 16B
      int id = tid * 2 + i;
      int r = id >> 2, c = id & 3;
      async_copy_b128(lds_off(&As[nb][r][c * 8]),
                      A + (size_t)(m0 + r) * lda + k0s + c * 8);
    }
    #pragma unroll
    for (int i = 0; i < 2; ++i) {              // B: 128 cols x 4 chunks of 16B
      int id = tid * 2 + i;
      int n = id >> 2, c = id & 3;
      async_copy_b128(lds_off(&Bs[nb][n][c * 4]),
                      Wp + ((size_t)(n0 + n) * K2 + (k0s >> 1) + c * 4) * 2);
    }
  };

  stage(0, 0);
  wait_async0();
  __syncthreads();

  int buf = 0;
  for (int k0 = 0; k0 < K; k0 += GBK) {
    const int cur = buf;
    const bool more = (k0 + GBK) < K;
    if (more) { buf ^= 1; stage(k0 + GBK, buf); }

    FragH fb[4];
    #pragma unroll
    for (int nt = 0; nt < 4; ++nt) {
      const unsigned* bp = &Bs[cur][wn * 64 + nt * 16 + (lane & 15)][khalf * 8];
      fb[nt].q[0] = *(const v4u*)(bp);
      fb[nt].q[1] = *(const v4u*)(bp + 4);
    }
    #pragma unroll
    for (int mt = 0; mt < 2; ++mt) {
      FragH fa;
      const _Float16* ap = &As[cur][wm * 32 + mt * 16 + mrow][khalf * 16];
      fa.q[0] = *(const v4u*)(ap);
      fa.q[1] = *(const v4u*)(ap + 8);
      #pragma unroll
      for (int nt = 0; nt < 4; ++nt)
        acc[mt][nt] = __builtin_amdgcn_wmma_f32_16x16x32_f16(
            false, fa.h, false, fb[nt].h, (short)0, acc[mt][nt], false, false);
    }

    if (more) wait_async0();   // our DMA into the next buffer must land pre-barrier
    __syncthreads();
  }

  // ---- epilogue (compile-time MODE): bias (+BN+ReLU), f32 or f16 stores ----
  const int n_lane  = lane & 15;
  const int m_half8 = (lane >> 4) * 8;
  #pragma unroll
  for (int mt = 0; mt < 2; ++mt) {
    #pragma unroll
    for (int nt = 0; nt < 4; ++nt) {
      const int col = n0 + wn * 64 + nt * 16 + n_lane;
      const float bv = bias[col];
      float sc = 1.f, bt = 0.f;
      if constexpr (MODE & 1) { sc = gamma[col] * rsqrtf(1.f + BN_EPS); bt = beta[col]; }
      #pragma unroll
      for (int r = 0; r < 8; ++r) {
        const int row = m0 + wm * 32 + mt * 16 + m_half8 + r;
        float v = acc[mt][nt][r] + bv;
        if constexpr (MODE & 1) v = fmaxf(v * sc + bt, 0.f);
        const size_t o = (size_t)row * ldc + col_off + col;
        if constexpr (MODE & 2) ((_Float16*)C)[o] = (_Float16)v;
        else                    ((float*)C)[o]    = v;
      }
    }
  }
}

// ---------------------------------------------------------------------------
// KNN: block per query; distances in LDS, 32 iterative argmin reductions.
// ---------------------------------------------------------------------------
__global__ __launch_bounds__(256) void knn_kernel(const float* __restrict__ src,
                                                  const float* __restrict__ tgt,
                                                  int* __restrict__ out) {
  __shared__ float d[NPTS];
  __shared__ float rmin[256];
  __shared__ int   rarg[256];
  const int q = blockIdx.x;
  const float qx = src[q * 3], qy = src[q * 3 + 1], qz = src[q * 3 + 2];
  for (int j = threadIdx.x; j < NPTS; j += 256) {
    float dx = qx - tgt[j * 3], dy = qy - tgt[j * 3 + 1], dz = qz - tgt[j * 3 + 2];
    d[j] = dx * dx + dy * dy + dz * dz;
  }
  __syncthreads();
  for (int s = 0; s < KNN_K; ++s) {
    float best = 3.0e38f; int bi = 0x7fffffff;
    for (int j = threadIdx.x; j < NPTS; j += 256) {
      float v = d[j];
      if (v < best || (v == best && j < bi)) { best = v; bi = j; }
    }
    rmin[threadIdx.x] = best; rarg[threadIdx.x] = bi;
    __syncthreads();
    for (int off = 128; off > 0; off >>= 1) {
      if (threadIdx.x < off) {
        float o = rmin[threadIdx.x + off]; int oi = rarg[threadIdx.x + off];
        float m = rmin[threadIdx.x];       int mi = rarg[threadIdx.x];
        if (o < m || (o == m && oi < mi)) { rmin[threadIdx.x] = o; rarg[threadIdx.x] = oi; }
      }
      __syncthreads();
    }
    if (threadIdx.x == 0) { out[q * KNN_K + s] = rarg[0]; d[rarg[0]] = 3.0e38f; }
    __syncthreads();
  }
}

// ---------------------------------------------------------------------------
// Local cross attention: block/query, wave = head, lane = neighbor.
// fp32 math, wave32 shuffle softmax; f16 output (feeds O-projection GEMM).
// ---------------------------------------------------------------------------
__global__ __launch_bounds__(256) void attn_kernel(const float* __restrict__ Q,
                                                   const float* __restrict__ Kb,
                                                   const float* __restrict__ Vb,
                                                   const int* __restrict__ idx,
                                                   _Float16* __restrict__ out) {
  const int q = blockIdx.x;
  const int h = threadIdx.x >> 5;
  const int j = threadIdx.x & 31;
  const int n = idx[q * KNN_K + j];
  const float* qp = Q  + (size_t)q * FDIM + h * 64;
  const float* kp = Kb + (size_t)n * FDIM + h * 64;
  float s = 0.f;
  #pragma unroll
  for (int d4 = 0; d4 < 16; ++d4) {
    float4 qq = *(const float4*)(qp + d4 * 4);
    float4 kk = *(const float4*)(kp + d4 * 4);
    s += qq.x * kk.x + qq.y * kk.y + qq.z * kk.z + qq.w * kk.w;
  }
  s *= 0.125f;
  float m = s;
  #pragma unroll
  for (int o = 16; o > 0; o >>= 1) m = fmaxf(m, __shfl_xor(m, o, 32));
  const float e = __expf(s - m);
  float sum = e;
  #pragma unroll
  for (int o = 16; o > 0; o >>= 1) sum += __shfl_xor(sum, o, 32);
  const float w = e / sum;
  const float* vp = Vb + (size_t)n * FDIM + h * 64;
  _Float16* op = out + (size_t)q * FDIM + h * 64;
  #pragma unroll
  for (int d4 = 0; d4 < 16; ++d4) {
    float4 vv = *(const float4*)(vp + d4 * 4);
    float rx = vv.x * w, ry = vv.y * w, rz = vv.z * w, rw = vv.w * w;
    #pragma unroll
    for (int o = 16; o > 0; o >>= 1) {
      rx += __shfl_xor(rx, o, 32);
      ry += __shfl_xor(ry, o, 32);
      rz += __shfl_xor(rz, o, 32);
      rw += __shfl_xor(rw, o, 32);
    }
    if (j == 0) {
      v4h r = {(_Float16)rx, (_Float16)ry, (_Float16)rz, (_Float16)rw};
      *(v4h*)(op + d4 * 4) = r;
    }
  }
}

// ---------------------------------------------------------------------------
// Column max over 8192 rows -> g[512]
// ---------------------------------------------------------------------------
__global__ __launch_bounds__(256) void maxpool_kernel(const float* __restrict__ X,
                                                      float* __restrict__ g) {
  int c = blockIdx.x * 256 + threadIdx.x;
  if (c >= FDIM) return;
  float m = -3.0e38f;
  for (int r = 0; r < NPTS; ++r) m = fmaxf(m, X[(size_t)r * FDIM + c]);
  g[c] = m;
}

// ---------------------------------------------------------------------------
// Pose head: 512 -> 256 -> 128 -> 6 (single block, fp32)
// ---------------------------------------------------------------------------
__global__ __launch_bounds__(256) void pose_kernel(const float* __restrict__ g,
                                                   const float* __restrict__ w1, const float* __restrict__ b1,
                                                   const float* __restrict__ w2, const float* __restrict__ b2,
                                                   const float* __restrict__ w3, const float* __restrict__ b3,
                                                   float* __restrict__ out) {
  __shared__ float h1[256];
  __shared__ float h2[128];
  const int t = threadIdx.x;
  float a = b1[t];
  for (int k = 0; k < 512; ++k) a += g[k] * w1[k * 256 + t];
  h1[t] = fmaxf(a, 0.f);
  __syncthreads();
  if (t < 128) {
    float a2 = b2[t];
    for (int k = 0; k < 256; ++k) a2 += h1[k] * w2[k * 128 + t];
    h2[t] = fmaxf(a2, 0.f);
  }
  __syncthreads();
  if (t < 6) {
    float a3 = b3[t];
    for (int k = 0; k < 128; ++k) a3 += h2[k] * w3[k * 6 + t];
    out[t] = a3;
  }
}

// ---------------------------------------------------------------------------
// Host orchestration. d_in layout = JAX pytree order (dict keys sorted):
//   0 source_points, 1 target_points,
//   attn L=0..2 @ base=2+8L: k.b,k.w,o.b,o.w,q.b,q.w,v.b,v.w
//   enc_s: 26 final.b, 27 final.w, layer i: 28+4i be, 29+4i g, 30+4i b, 31+4i w
//   enc_t: 40..53 same shape
//   fusion: 54 b1, 55 b2, 56 g1, 57 g2, 58 l1.b, 59 l1.w, 60 l2.b, 61 l2.w
//   pose: 62 l1.b, 63 l1.w, 64 l2.b, 65 l2.w, 66 l3.b, 67 l3.w
// ---------------------------------------------------------------------------
extern "C" void kernel_launch(void* const* d_in, const int* in_sizes, int n_in,
                              void* d_out, int out_size, void* d_ws, size_t ws_size,
                              hipStream_t stream) {
  (void)in_sizes; (void)n_in; (void)out_size; (void)ws_size;
  const float* src = (const float*)d_in[0];
  const float* tgt = (const float*)d_in[1];
  #define P(i) ((const float*)d_in[(i)])

  const size_t MB = 1ull << 20;
  char* ws = (char*)d_ws;
  float*    qbuf   = (float*)(ws +  0 * MB);      // 8192x512 f32 (Q proj)
  float*    kbuf   = (float*)(ws + 16 * MB);      // 8192x512 f32 (K proj / fusion out)
  float*    vbuf   = (float*)(ws + 32 * MB);      // 8192x512 f32 (V proj)
  _Float16* xcat16 = (_Float16*)(ws + 48 * MB);   // 8192x1024 f16: [sf0 | sf_cur]
  _Float16* tf16   = (_Float16*)(ws + 64 * MB);   // 8192x512 f16
  _Float16* ao16   = (_Float16*)(ws + 72 * MB);   // 8192x512 f16 attention out
  _Float16* fu16   = (_Float16*)(ws + 80 * MB);   // 8192x512 f16 fusion l1 out
  _Float16* a16a   = (_Float16*)(ws + 88 * MB);   // 8192x64
  _Float16* a16b   = (_Float16*)(ws + 90 * MB);   // 8192x128
  _Float16* a16c   = (_Float16*)(ws + 92 * MB);   // 8192x256
  int*      idxb   = (int*)    (ws + 96 * MB);    // 8192x32
  float*    gbuf   = (float*)  (ws + 97 * MB);    // 512
  _Float16* w16    = (_Float16*)(ws + 98 * MB);   // ~8.6MB packed f16 weights

  // ---- weight prep: f32 -> f16 N-major K-pair-packed ----
  size_t cur = 0;
  auto cvtp = [&](int pi, int K, int N) -> _Float16* {
    _Float16* dst = w16 + cur;
    int tot = (K / 2) * N;
    cvt_pack_kernel<<<(tot + 255) / 256, 256, 0, stream>>>(P(pi), dst, K, N);
    cur += (size_t)K * N;
    return dst;
  };
  _Float16* w_s1 = cvtp(31,   64, 128);
  _Float16* w_s2 = cvtp(35,  128, 256);
  _Float16* w_sf = cvtp(27,  256, 512);
  _Float16* w_t1 = cvtp(45,   64, 128);
  _Float16* w_t2 = cvtp(49,  128, 256);
  _Float16* w_tf = cvtp(41,  256, 512);
  _Float16 *w_q[3], *w_k[3], *w_v[3], *w_o[3];
  for (int L = 0; L < 3; ++L) {
    const int base = 2 + 8 * L;
    w_k[L] = cvtp(base + 1, 512, 512);
    w_o[L] = cvtp(base + 3, 512, 512);
    w_q[L] = cvtp(base + 5, 512, 512);
    w_v[L] = cvtp(base + 7, 512, 512);
  }
  _Float16* w_f1 = cvtp(59, 1024, 512);
  _Float16* w_f2 = cvtp(61,  512, 512);

  auto gemm = [&](const _Float16* A, int lda, const _Float16* W, const float* bias,
                  const float* g, const float* be, void* C, int ldc, int coloff,
                  int N, int K, int mode) {
    dim3 grid(NPTS / GBM, N / GBN);
    switch (mode) {
      case 0: gemm_wmma_kernel<0><<<grid, 256, 0, stream>>>(A, lda, W, bias, g, be, C, ldc, coloff, N, K); break;
      case 1: gemm_wmma_kernel<1><<<grid, 256, 0, stream>>>(A, lda, W, bias, g, be, C, ldc, coloff, N, K); break;
      case 2: gemm_wmma_kernel<2><<<grid, 256, 0, stream>>>(A, lda, W, bias, g, be, C, ldc, coloff, N, K); break;
      default:gemm_wmma_kernel<3><<<grid, 256, 0, stream>>>(A, lda, W, bias, g, be, C, ldc, coloff, N, K); break;
    }
  };

  // ---- encoders (f16 end-to-end) ----
  enc0_kernel<<<NPTS / 4, 256, 0, stream>>>(src, P(31), P(30), P(29), P(28), a16a);
  gemm(a16a,  64, w_s1, P(34), P(33), P(32), a16b, 128, 0, 128,  64, 3);
  gemm(a16b, 128, w_s2, P(38), P(37), P(36), a16c, 256, 0, 256, 128, 3);
  gemm(a16c, 256, w_sf, P(26), nullptr, nullptr, xcat16, 1024, 0, 512, 256, 2); // sf0

  enc0_kernel<<<NPTS / 4, 256, 0, stream>>>(tgt, P(45), P(44), P(43), P(42), a16a);
  gemm(a16a,  64, w_t1, P(48), P(47), P(46), a16b, 128, 0, 128,  64, 3);
  gemm(a16b, 128, w_t2, P(52), P(51), P(50), a16c, 256, 0, 256, 128, 3);
  gemm(a16c, 256, w_tf, P(40), nullptr, nullptr, tf16, 512, 0, 512, 256, 2);

  // ---- KNN ----
  knn_kernel<<<NPTS, 256, 0, stream>>>(src, tgt, idxb);

  // ---- 3 cross-attention layers; sf_cur lives in xcat16[:, 512:] ----
  for (int L = 0; L < 3; ++L) {
    const int base = 2 + 8 * L;
    const _Float16* sfA = (L == 0) ? xcat16 : (xcat16 + 512);
    gemm(sfA,  1024, w_q[L], P(base + 4), nullptr, nullptr, qbuf, 512, 0, 512, 512, 0);
    gemm(tf16,  512, w_k[L], P(base + 0), nullptr, nullptr, kbuf, 512, 0, 512, 512, 0);
    gemm(tf16,  512, w_v[L], P(base + 6), nullptr, nullptr, vbuf, 512, 0, 512, 512, 0);
    attn_kernel<<<NPTS, 256, 0, stream>>>(qbuf, kbuf, vbuf, idxb, ao16);
    gemm(ao16,  512, w_o[L], P(base + 2), nullptr, nullptr, xcat16, 1024, 512, 512, 512, 2);
  }

  // ---- fusion on concat (already contiguous f16) + pool + pose head ----
  gemm(xcat16, 1024, w_f1, P(58), P(56), P(54), fu16, 512, 0, 512, 1024, 3);
  gemm(fu16,    512, w_f2, P(60), P(57), P(55), kbuf, 512, 0, 512,  512, 1);
  maxpool_kernel<<<2, 256, 0, stream>>>(kbuf, gbuf);
  pose_kernel<<<1, 256, 0, stream>>>(gbuf, P(63), P(62), P(65), P(64), P(67), P(66),
                                     (float*)d_out);
  #undef P
}
